// Attention_8263517077742
// MI455X (gfx1250) — matrix operand order repaired and checked by reading, not compiled, not run on hardware
//
#include <hip/hip_runtime.h>

// ---------------------------------------------------------------- dims
#define B_     2
#define G_     2
#define N_     2048
#define D_     512
#define H_     8
#define DH_    64
#define DI_    512           // H*DH
#define BG_    (B_*G_)       // 4
#define BGH_   (BG_*H_)      // 32
#define NKPAD_ 2080          // 65 * 32 key tiles (key 0 = null kv, 1..2048 real, rest pad)
#define NEG_BIG (-1e30f)

typedef __attribute__((ext_vector_type(16))) _Float16 v16h;
typedef __attribute__((ext_vector_type(8)))  _Float16 v8h;
typedef __attribute__((ext_vector_type(4)))  _Float16 v4h;
typedef __attribute__((ext_vector_type(8)))  float    v8f;

union V16 { v16h v; v8h h[2]; };

// ---------------------------------------------------------------- WMMA GEMM core
// C(16x16,f32) += A(16x32,f16) x B(32x16,f16), looped over K.
// A row-major [16 rows][lda], B row-major [16 cols][ldb] (B holds the
// transposed operand: lane's column reads K-dim contiguously).
static __device__ inline v8f wmma_tile_gemm(const _Float16* __restrict__ A, int lda,
                                            const _Float16* __restrict__ Bm, int ldb,
                                            int K, int lane) {
  const int col  = lane & 15;
  const int hsel = (lane >> 4) & 1;
  const _Float16* ap = A + (size_t)col * lda;
  const _Float16* bp = Bm + (size_t)col * ldb;
  v8f acc = {};
  for (int k = 0; k < K; k += 32) {
    V16 a, b;
    // A 16x32 layout: lanes 0-15 hold K 0..15, lanes 16-31 hold K 16..31 (contiguous)
    a.h[0] = *(const v8h*)(ap + k + hsel * 16);
    a.h[1] = *(const v8h*)(ap + k + hsel * 16 + 8);
    // B 32x16 layout: lanes 0-15 hold K 0..15, lanes 16-31 hold K 16..31 (contiguous)
    b.h[0] = *(const v8h*)(bp + k + hsel * 16);
    b.h[1] = *(const v8h*)(bp + k + hsel * 16 + 8);
    acc = __builtin_amdgcn_wmma_f32_16x16x32_f16(false, a.v, false, b.v,
                                                 (short)0, acc, false, false);
  }
  return acc;
}

// ---------------------------------------------------------------- prep kernels
// Weights -> f16 with gamma folded in (w[g][e][d] * gamma[g][d]).
__global__ void prep_w_gamma(const float* __restrict__ w, const float* __restrict__ gamma,
                             _Float16* __restrict__ wh, size_t total, int gshift) {
  for (size_t i = (size_t)blockIdx.x * blockDim.x + threadIdx.x; i < total;
       i += (size_t)gridDim.x * blockDim.x) {
    int d = (int)(i & (D_ - 1));
    int g = (int)(i >> gshift);
    wh[i] = (_Float16)(w[i] * gamma[g * D_ + d]);
  }
}
__global__ void prep_w_plain(const float* __restrict__ w, _Float16* __restrict__ wh,
                             size_t total) {
  for (size_t i = (size_t)blockIdx.x * blockDim.x + threadIdx.x; i < total;
       i += (size_t)gridDim.x * blockDim.x)
    wh[i] = (_Float16)w[i];
}

// Per row: f16 copy of x + rowscale = sqrt(D)/max(||x||,eps). One wave per row.
__global__ __launch_bounds__(256) void prep_x_kernel(const float* __restrict__ x,
                                                     _Float16* __restrict__ xh,
                                                     float* __restrict__ rowscale) {
  int wave = threadIdx.x >> 5, lane = threadIdx.x & 31;
  size_t row = (size_t)blockIdx.x * 8 + wave;      // BG_*N_ rows
  const float4* xr = (const float4*)(x + row * D_);
  _Float16* xo = xh + row * D_;
  float ss = 0.f;
#pragma unroll
  for (int i = 0; i < 4; ++i) {
    float4 v = xr[lane + i * 32];
    ss += v.x * v.x + v.y * v.y + v.z * v.z + v.w * v.w;
    v4h hv = { (_Float16)v.x, (_Float16)v.y, (_Float16)v.z, (_Float16)v.w };
    *(v4h*)(xo + (lane + i * 32) * 4) = hv;
  }
#pragma unroll
  for (int m = 16; m >= 1; m >>= 1) ss += __shfl_xor(ss, m, 32);
  if (lane == 0) rowscale[row] = sqrtf((float)D_) / fmaxf(sqrtf(ss), 1e-12f);
}

// Null-kv row 0 + zero pad rows/cols of K and V^T.
__global__ void prep_nullpad(const float* __restrict__ null_kv,
                             _Float16* __restrict__ kh, _Float16* __restrict__ vT) {
  int bgh = blockIdx.x;                  // 0..31
  int g = (bgh >> 3) & (G_ - 1), h = bgh & (H_ - 1);
  int t = threadIdx.x;
  if (t < DH_) {
    float nk = null_kv[((0 * G_ + g) * H_ + h) * DH_ + t];
    float nv = null_kv[((1 * G_ + g) * H_ + h) * DH_ + t];
    kh[(size_t)bgh * NKPAD_ * DH_ + t] = (_Float16)nk;
    vT[((size_t)bgh * DH_ + t) * NKPAD_ + 0] = (_Float16)nv;
  }
  for (int i = t; i < (NKPAD_ - (N_ + 1)) * DH_; i += blockDim.x) {
    int rr = i >> 6, cc = i & 63;
    kh[((size_t)bgh * NKPAD_ + (N_ + 1) + rr) * DH_ + cc] = (_Float16)0.f;
    vT[((size_t)bgh * DH_ + cc) * NKPAD_ + (N_ + 1) + rr] = (_Float16)0.f;
  }
}

__global__ void prep_bias(const int* __restrict__ mask, float* __restrict__ bias) {
  int b = blockIdx.x;
  for (int j = threadIdx.x; j < NKPAD_; j += blockDim.x) {
    float v;
    if (j == 0) v = 0.f;
    else if (j <= N_) v = mask[b * N_ + (j - 1)] ? 0.f : NEG_BIG;
    else v = NEG_BIG;
    bias[b * NKPAD_ + j] = v;
  }
}

// ---------------------------------------------------------------- projections
// q[b,g,h,n,dh] = 0.125 * rowscale * (x @ (wq*gamma)^T)
__global__ __launch_bounds__(256) void proj_q_kernel(const _Float16* __restrict__ xh,
                                                     const _Float16* __restrict__ wqh,
                                                     const float* __restrict__ rowscale,
                                                     _Float16* __restrict__ qh) {
  int lane = threadIdx.x & 31, wave = threadIdx.x >> 5;
  int col = lane & 15, hsel = (lane >> 4) & 1;
  int tile = blockIdx.x * 8 + wave;       // 128 m-tiles * 32 e-tiles
  int te = tile & 31, tm = tile >> 5;
  int bg = blockIdx.y, g = bg & (G_ - 1);
  int m0 = tm * 16, e0 = te * 16;
  v8f acc = wmma_tile_gemm(xh + ((size_t)bg * N_ + m0) * D_, D_,
                           wqh + ((size_t)g * DI_ + e0) * D_, D_, D_, lane);
  int e = e0 + col, h = e >> 6, dh = e & 63;
#pragma unroll
  for (int r = 0; r < 8; ++r) {
    int n = m0 + r + hsel * 8;
    float v = acc[r] * rowscale[bg * N_ + n] * 0.125f;   // DH^-0.5 folded here
    qh[(((size_t)bg * H_ + h) * N_ + n) * DH_ + dh] = (_Float16)v;
  }
}

// kv: k -> kh[b,g,h,1+n,dh], v -> vT[b,g,h,dh,1+n]
__global__ __launch_bounds__(256) void proj_kv_kernel(const _Float16* __restrict__ xh,
                                                      const _Float16* __restrict__ wkvh,
                                                      const float* __restrict__ rowscale,
                                                      _Float16* __restrict__ kh,
                                                      _Float16* __restrict__ vT) {
  int lane = threadIdx.x & 31, wave = threadIdx.x >> 5;
  int col = lane & 15, hsel = (lane >> 4) & 1;
  int tile = blockIdx.x * 8 + wave;       // 128 m-tiles * 64 e-tiles
  int te = tile & 63, tm = tile >> 6;
  int bg = blockIdx.y, g = bg & (G_ - 1);
  int m0 = tm * 16, e0 = te * 16;
  v8f acc = wmma_tile_gemm(xh + ((size_t)bg * N_ + m0) * D_, D_,
                           wkvh + ((size_t)g * 2 * DI_ + e0) * D_, D_, D_, lane);
  int e = e0 + col;
#pragma unroll
  for (int r = 0; r < 8; ++r) {
    int n = m0 + r + hsel * 8;
    float v = acc[r] * rowscale[bg * N_ + n];
    if (e < DI_) {
      int h = e >> 6, dh = e & 63;
      kh[(((size_t)bg * H_ + h) * NKPAD_ + n + 1) * DH_ + dh] = (_Float16)v;
    } else {
      int ev = e - DI_, h = ev >> 6, dh = ev & 63;
      vT[(((size_t)bg * H_ + h) * DH_ + dh) * NKPAD_ + n + 1] = (_Float16)v;
    }
  }
}

// ---------------------------------------------------------------- flash attention
// One wave = 16 query rows x all 2080 keys, 32-key chunks, online softmax.
__global__ __launch_bounds__(256) void attn_kernel(const _Float16* __restrict__ qh,
                                                   const _Float16* __restrict__ kh,
                                                   const _Float16* __restrict__ vT,
                                                   const float* __restrict__ bias,
                                                   _Float16* __restrict__ att) {
  __shared__ __align__(16) _Float16 lds[8][16][32];   // per-wave P patch (C->A layout)
  int wave = threadIdx.x >> 5, lane = threadIdx.x & 31;
  int col = lane & 15, hsel = (lane >> 4) & 1;
  int bgh = blockIdx.y;
  int bg = bgh >> 3, h = bgh & 7, b = bg >> 1;
  int m0 = (blockIdx.x * 8 + wave) * 16;

  // Q A-operands (loaded once; reused across all 65 key iterations)
  const _Float16* qrow = qh + ((size_t)bgh * N_ + m0 + col) * DH_;
  V16 aql, aqu;
  aql.h[0] = *(const v8h*)(qrow + hsel * 16);
  aql.h[1] = *(const v8h*)(qrow + hsel * 16 + 8);
  aqu.h[0] = *(const v8h*)(qrow + 32 + hsel * 16);
  aqu.h[1] = *(const v8h*)(qrow + 32 + hsel * 16 + 8);

  const _Float16* kb = kh + (size_t)bgh * NKPAD_ * DH_;
  const _Float16* vb = vT + (size_t)bgh * DH_ * NKPAD_;
  const float* bb = bias + b * NKPAD_;

  float m_r[8], l_r[8];
  v8f O[4];
#pragma unroll
  for (int r = 0; r < 8; ++r) { m_r[r] = NEG_BIG; l_r[r] = 0.f; }
#pragma unroll
  for (int t = 0; t < 4; ++t) O[t] = (v8f){};

  for (int j0 = 0; j0 < NKPAD_; j0 += 32) {
    // --- scores: two 16-key tiles, K-dim = DH split into 2x32
    const _Float16* kr0 = kb + (size_t)(j0 + col) * DH_;
    const _Float16* kr1 = kb + (size_t)(j0 + 16 + col) * DH_;
    V16 b00, b01, b10, b11;
    b00.h[0] = *(const v8h*)(kr0 + hsel * 16);      b00.h[1] = *(const v8h*)(kr0 + hsel * 16 + 8);
    b01.h[0] = *(const v8h*)(kr0 + 32 + hsel * 16); b01.h[1] = *(const v8h*)(kr0 + 32 + hsel * 16 + 8);
    b10.h[0] = *(const v8h*)(kr1 + hsel * 16);      b10.h[1] = *(const v8h*)(kr1 + hsel * 16 + 8);
    b11.h[0] = *(const v8h*)(kr1 + 32 + hsel * 16); b11.h[1] = *(const v8h*)(kr1 + 32 + hsel * 16 + 8);
    v8f S0 = {}, S1 = {};
    S0 = __builtin_amdgcn_wmma_f32_16x16x32_f16(false, aql.v, false, b00.v, (short)0, S0, false, false);
    S0 = __builtin_amdgcn_wmma_f32_16x16x32_f16(false, aqu.v, false, b01.v, (short)0, S0, false, false);
    S1 = __builtin_amdgcn_wmma_f32_16x16x32_f16(false, aql.v, false, b10.v, (short)0, S1, false, false);
    S1 = __builtin_amdgcn_wmma_f32_16x16x32_f16(false, aqu.v, false, b11.v, (short)0, S1, false, false);

    float bias0 = bb[j0 + col];
    float bias1 = bb[j0 + 16 + col];

    // --- online softmax per row (row lives across 16 lanes of one half-wave)
#pragma unroll
    for (int r = 0; r < 8; ++r) {
      float s0 = S0[r] + bias0;
      float s1 = S1[r] + bias1;
      float mr = fmaxf(s0, s1);
      mr = fmaxf(mr, __shfl_xor(mr, 1, 32));
      mr = fmaxf(mr, __shfl_xor(mr, 2, 32));
      mr = fmaxf(mr, __shfl_xor(mr, 4, 32));
      mr = fmaxf(mr, __shfl_xor(mr, 8, 32));
      float mn = fmaxf(m_r[r], mr);
      float alpha = __expf(m_r[r] - mn);
      m_r[r] = mn;
      float p0 = __expf(s0 - mn);
      float p1 = __expf(s1 - mn);
      float rs = p0 + p1;
      rs += __shfl_xor(rs, 1, 32);
      rs += __shfl_xor(rs, 2, 32);
      rs += __shfl_xor(rs, 4, 32);
      rs += __shfl_xor(rs, 8, 32);
      l_r[r] = l_r[r] * alpha + rs;
      int row = r + hsel * 8;
      lds[wave][row][col]      = (_Float16)p0;   // stage P in [row][key] order
      lds[wave][row][col + 16] = (_Float16)p1;
#pragma unroll
      for (int t = 0; t < 4; ++t) O[t][r] *= alpha;
    }

    // --- P as A-operand (16x32 over keys) from LDS
    V16 pa;
    pa.h[0] = *(const v8h*)(&lds[wave][col][hsel * 16]);
    pa.h[1] = *(const v8h*)(&lds[wave][col][hsel * 16 + 8]);

    // --- O += P x V ; V^T gives contiguous key runs per dh column
#pragma unroll
    for (int t = 0; t < 4; ++t) {
      const _Float16* vr = vb + (size_t)(t * 16 + col) * NKPAD_ + j0 + hsel * 16;
      V16 bv;
      bv.h[0] = *(const v8h*)(vr);
      bv.h[1] = *(const v8h*)(vr + 8);
      O[t] = __builtin_amdgcn_wmma_f32_16x16x32_f16(false, pa.v, false, bv.v, (short)0, O[t], false, false);
    }
  }

  // --- epilogue: normalize, merge heads into att[bg][n][h*64+dh] (f16)
#pragma unroll
  for (int r = 0; r < 8; ++r) {
    float inv = 1.f / l_r[r];
    int n = m0 + r + hsel * 8;
    size_t base = ((size_t)bg * N_ + n) * DI_ + h * DH_ + col;
#pragma unroll
    for (int t = 0; t < 4; ++t)
      att[base + t * 16] = (_Float16)(O[t][r] * inv);
  }
}

// ---------------------------------------------------------------- output projection
__global__ __launch_bounds__(256) void proj_out_kernel(const _Float16* __restrict__ att,
                                                       const _Float16* __restrict__ wouth,
                                                       float* __restrict__ out) {
  int lane = threadIdx.x & 31, wave = threadIdx.x >> 5;
  int col = lane & 15, hsel = (lane >> 4) & 1;
  int tile = blockIdx.x * 8 + wave;       // 128 m-tiles * 32 d-tiles
  int td = tile & 31, tm = tile >> 5;
  int bg = blockIdx.y, g = bg & (G_ - 1);
  int m0 = tm * 16, d0 = td * 16;
  v8f acc = wmma_tile_gemm(att + ((size_t)bg * N_ + m0) * DI_, DI_,
                           wouth + ((size_t)g * D_ + d0) * DI_, DI_, DI_, lane);
#pragma unroll
  for (int r = 0; r < 8; ++r) {
    int n = m0 + r + hsel * 8;
    out[((size_t)bg * N_ + n) * D_ + d0 + col] = acc[r];
  }
}

// ---------------------------------------------------------------- launch
extern "C" void kernel_launch(void* const* d_in, const int* in_sizes, int n_in,
                              void* d_out, int out_size, void* d_ws, size_t ws_size,
                              hipStream_t stream) {
  (void)in_sizes; (void)n_in; (void)out_size; (void)ws_size;
  const float* x        = (const float*)d_in[0];
  const int*   mask     = (const int*)d_in[1];
  const float* gamma_q  = (const float*)d_in[2];
  const float* gamma_c  = (const float*)d_in[3];
  const float* wq       = (const float*)d_in[4];
  const float* wkv      = (const float*)d_in[5];
  const float* wout     = (const float*)d_in[6];
  const float* null_kv  = (const float*)d_in[7];
  float* out = (float*)d_out;

  char* ws = (char*)d_ws;
  size_t off = 0;
  auto take = [&](size_t bytes) { char* p = ws + off; off += (bytes + 255) & ~(size_t)255; return p; };

  _Float16* xh     = (_Float16*)take((size_t)BG_ * N_ * D_ * 2);
  float*    rscale = (float*)   take((size_t)BG_ * N_ * 4);
  _Float16* wqh    = (_Float16*)take((size_t)G_ * DI_ * D_ * 2);
  _Float16* wkvh   = (_Float16*)take((size_t)G_ * 2 * DI_ * D_ * 2);
  _Float16* wouth  = (_Float16*)take((size_t)G_ * D_ * DI_ * 2);
  _Float16* qh     = (_Float16*)take((size_t)BGH_ * N_ * DH_ * 2);
  _Float16* kh     = (_Float16*)take((size_t)BGH_ * NKPAD_ * DH_ * 2);
  _Float16* vT     = (_Float16*)take((size_t)BGH_ * DH_ * NKPAD_ * 2);
  float*    bias   = (float*)   take((size_t)B_ * NKPAD_ * 4);
  _Float16* att    = xh;  // alias: xh fully consumed before attn writes att

  // weights (gamma folded)
  prep_w_gamma<<<1024, 256, 0, stream>>>(wq,  gamma_q, wqh,  (size_t)G_ * DI_ * D_, 18);
  prep_w_gamma<<<1024, 256, 0, stream>>>(wkv, gamma_c, wkvh, (size_t)G_ * 2 * DI_ * D_, 19);
  prep_w_plain<<<1024, 256, 0, stream>>>(wout, wouth, (size_t)G_ * D_ * DI_);
  // x -> f16 + row scales
  prep_x_kernel<<<(BG_ * N_) / 8, 256, 0, stream>>>(x, xh, rscale);
  // null kv + padding + mask bias
  prep_nullpad<<<BGH_, 256, 0, stream>>>(null_kv, kh, vT);
  prep_bias<<<B_, 256, 0, stream>>>(mask, bias);
  // projections
  proj_q_kernel <<<dim3((N_ / 16) * (DI_ / 16) / 8, BG_), 256, 0, stream>>>(xh, wqh, rscale, qh);
  proj_kv_kernel<<<dim3((N_ / 16) * (2 * DI_ / 16) / 8, BG_), 256, 0, stream>>>(xh, wkvh, rscale, kh, vT);
  // flash attention (writes att, aliasing xh after its last read)
  attn_kernel<<<dim3(N_ / (16 * 8), BGH_), 256, 0, stream>>>(qh, kh, vT, bias, att);
  // output projection
  proj_out_kernel<<<dim3((N_ / 16) * (D_ / 16) / 8, BG_), 256, 0, stream>>>(att, wouth, out);
}
